// CrossModalAttention_34978213658699
// MI455X (gfx1250) — compile-verified, matrix operand order checked by reading, not run on hardware
//
#include <hip/hip_runtime.h>
#include <hip/hip_bf16.h>
#include <hip/hip_fp16.h>

// Problem constants (from reference)
#define B_    16
#define S_    1024
#define VDIM_ 768
#define TDIM_ 768
#define HID_  1024
#define NH_   16
#define HD_   64
#define L_    1025          // S + 1 (vision token prepended)
#define LPAD_ 1056          // pad L to multiple of 32 for WMMA tiling
#define MT_   (LPAD_ / 16)  // 66 row-tiles

typedef __attribute__((ext_vector_type(16))) _Float16 v16h;
typedef __attribute__((ext_vector_type(8)))  _Float16 v8h;
typedef __attribute__((ext_vector_type(8)))  float    v8f;
typedef __attribute__((ext_vector_type(4)))  unsigned int v4u;
typedef __attribute__((ext_vector_type(8)))  int      v8i;
typedef __attribute__((ext_vector_type(4)))  int      v4i;

// ---------------------------------------------------------------------------
// WMMA fragment loaders (wave32; 16-bit A/B 16x32 layout per CDNA5 ISA 7.12.2:
// lane L holds row M=L&15; lanes<16 K-groups {0..7,16..23}, lanes>=16 {8..15,24..31})
// ---------------------------------------------------------------------------
__device__ __forceinline__ v16h frag_from_f16(const _Float16* __restrict__ P,
                                              int ld, int row, int k0, int lane) {
  const int g = (lane & 16) ? 8 : 0;
  const _Float16* p = P + (size_t)row * ld + k0 + g;
  v8h lo = *(const v8h*)p;
  v8h hi = *(const v8h*)(p + 16);
  return __builtin_shufflevector(lo, hi, 0,1,2,3,4,5,6,7,8,9,10,11,12,13,14,15);
}

__device__ __forceinline__ v16h frag_from_f32(const float* __restrict__ P,
                                              int ld, int row, int k0, int lane) {
  const int g = (lane & 16) ? 8 : 0;
  const float* p = P + (size_t)row * ld + k0 + g;
  v16h r;
#pragma unroll
  for (int i = 0; i < 8; ++i) {
    r[i]     = (_Float16)p[i];
    r[8 + i] = (_Float16)p[16 + i];
  }
  return r;
}

// ---------------------------------------------------------------------------
// Tensor Data Mover: async-load a 2D f16 tile (rows x kelems) from global into
// LDS (contiguous rows of kelems). D# built per CDNA5 ISA 08_async_tensor §8.
// Group0: count=1 | lds_addr | global_addr | type=2
// Group1: data_size=2B | tensor dims | tile dims | dim0 stride (row pitch)
// 6-arg builtin form (this toolchain): (g0, g1, g2, g3, g4, cpol)
// ---------------------------------------------------------------------------
__device__ __forceinline__ void tdm_load_tile_f16(unsigned lds_off,
                                                  const void* gptr,
                                                  unsigned kelems,     // tile_dim0
                                                  unsigned rows,       // tile_dim1
                                                  unsigned pitch) {    // row pitch (elems)
  unsigned long long ga = (unsigned long long)(uintptr_t)gptr;
  v4u g0;
  g0[0] = 1u;                                          // count=1 (valid descriptor)
  g0[1] = lds_off;                                     // lds_addr (bytes)
  g0[2] = (unsigned)(ga & 0xFFFFFFFFu);                // global_addr[31:0]
  g0[3] = (unsigned)((ga >> 32) & 0x01FFFFFFu) | (2u << 30);  // addr[56:32] | type=2
  v8i g1;
  g1[0] = (int)(1u << 16);                             // data_size=1 -> 2 bytes
  g1[1] = (int)((kelems & 0xFFFFu) << 16);             // tensor_dim0[15:0]
  g1[2] = (int)((kelems >> 16) | ((rows & 0xFFFFu) << 16));   // dim0 hi | tensor_dim1 lo
  g1[3] = (int)((rows >> 16) | ((kelems & 0xFFFFu) << 16));   // dim1 hi | tile_dim0
  g1[4] = (int)(rows & 0xFFFFu);                       // tile_dim1 (tile_dim2 = 0)
  g1[5] = (int)pitch;                                  // tensor_dim0_stride[31:0]
  g1[6] = 0;                                           // stride0 hi | stride1 lo
  g1[7] = 0;                                           // stride1 hi
  v4i z4 = {0, 0, 0, 0};                               // groups 2/3 unused (2D)
  v8i z8 = {0, 0, 0, 0, 0, 0, 0, 0};                   // trailing group unused
  __builtin_amdgcn_tensor_load_to_lds(g0, g1, z4, z4, z8, 0);
}

// ---------------------------------------------------------------------------
// Workspace init: zero abs-sum accumulators and padded rows of combined (f32+f16)
// ---------------------------------------------------------------------------
__global__ void init_ws_kernel(float* __restrict__ sums, float* __restrict__ comb,
                               _Float16* __restrict__ comb_h) {
  int idx = blockIdx.x * blockDim.x + threadIdx.x;
  if (idx < 8) sums[idx] = 0.f;
  const int padrows = LPAD_ - L_;            // 31
  const int total   = B_ * padrows * HID_;
  for (int i = idx; i < total; i += gridDim.x * blockDim.x) {
    int b   = i / (padrows * HID_);
    int rem = i % (padrows * HID_);
    int r   = rem / HID_;
    int c   = rem % HID_;
    size_t o = ((size_t)b * LPAD_ + L_ + r) * HID_ + c;
    comb[o]   = 0.f;
    comb_h[o] = (_Float16)0.f;
  }
}

// Precompute additive attention bias per (batch, key column)
__global__ void attn_bias_kernel(const int* __restrict__ mask, float* __restrict__ attb) {
  int i = blockIdx.x * blockDim.x + threadIdx.x;
  if (i >= B_ * LPAD_) return;
  int b = i / LPAD_, j = i % LPAD_;
  float v;
  if (j == 0)       v = 0.f;
  else if (j <= S_) v = (1.f - (float)mask[b * S_ + j - 1]) * -10000.f;
  else              v = -1e30f;     // exclude padded keys
  attb[i] = v;
}

// ---------------------------------------------------------------------------
// sum(|W|) reduction -> single float via atomicAdd
// ---------------------------------------------------------------------------
__global__ __launch_bounds__(256) void abs_sum_kernel(const float* __restrict__ W,
                                                      int n, float* __restrict__ out) {
  __shared__ float red[256];
  float s = 0.f;
  for (int i = blockIdx.x * 256 + threadIdx.x; i < n; i += gridDim.x * 256)
    s += fabsf(W[i]);
  red[threadIdx.x] = s;
  __syncthreads();
  for (int off = 128; off > 0; off >>= 1) {
    if (threadIdx.x < off) red[threadIdx.x] += red[threadIdx.x + off];
    __syncthreads();
  }
  if (threadIdx.x == 0) atomicAdd(out, red[0]);
}

// ---------------------------------------------------------------------------
// Ternary-quantize weights into f16 {-1,0,+1} (exact in f16; scale applied in
// the GEMM epilogue: x @ (q*s)^T = (x @ q^T) * s)
// ---------------------------------------------------------------------------
__global__ __launch_bounds__(256) void quantize_kernel(const float* __restrict__ W, int n,
                                                       const float* __restrict__ sum,
                                                       float inv_n,
                                                       _Float16* __restrict__ qw) {
  float s    = fminf(fmaxf(sum[0] * inv_n, 1e-5f), 1000.0f);
  float invs = 1.0f / s;
  for (int i = blockIdx.x * blockDim.x + threadIdx.x; i < n;
       i += gridDim.x * blockDim.x) {
    float wn = fminf(fmaxf(W[i] * invs, -10.f), 10.f);
    float q  = (wn > 0.6666667f) ? 1.f : ((wn < -0.6666667f) ? -1.f : 0.f);
    qw[i] = (_Float16)q;
  }
}

// ---------------------------------------------------------------------------
// Bit-linear GEMM, TDM-staged weights.
// Workgroup (8 waves) computes 128 rows x 64 cols. Per 32-wide K-step, wave 0
// DMAs the 64x32 f16 weight strip into LDS (tensor_load_to_lds, TENSORcnt),
// then every wave runs 4 WMMAs against it (A frag loaded once, reused 4x).
// OUT_MODE: 0 = f32, 1 = f16, 2 = f16 transposed per-head (for V),
//           3 = f32 + f16 shadow.
// Output-row remap: orow = m + (m / rows_per_batch) * row_pad + row_base.
// ---------------------------------------------------------------------------
template <bool A_F16, int OUT_MODE>
__global__ __launch_bounds__(256) void gemm_bitlinear(
    const void* __restrict__ Ain, const _Float16* __restrict__ Qw,
    const float* __restrict__ bias, const float* __restrict__ abs_sum,
    float inv_numel, void* __restrict__ Out, void* __restrict__ Out2,
    int M, int N, int K, int rows_per_batch, int row_pad, int row_base,
    int out_ld) {
  __shared__ _Float16 Bt[64][32];            // weight strip: rows n0..n0+63, k0..k0+31

  const int lane    = threadIdx.x & 31;
  const int wave    = threadIdx.x >> 5;
  const int mtiles  = (M + 15) >> 4;
  const int mblocks = (mtiles + 7) >> 3;
  const int bm = blockIdx.x % mblocks;
  const int ns = blockIdx.x / mblocks;
  const int m0 = (bm * 8 + wave) * 16;
  const int n0 = ns * 64;
  const int arow = min(m0 + (lane & 15), M - 1);  // clamp; stores are guarded
  const int g    = (lane & 16) ? 8 : 0;

  v8f acc[4] = {};
  for (int k0 = 0; k0 < K; k0 += 32) {
    __syncthreads();                         // previous strip fully consumed
    if (wave == 0) {
      tdm_load_tile_f16((unsigned)(uintptr_t)&Bt[0][0],
                        Qw + (size_t)n0 * K + k0,
                        32u, 64u, (unsigned)K);
      __builtin_amdgcn_s_wait_tensorcnt(0);
    }
    __syncthreads();                         // strip visible to all waves

    if (A_F16 && k0 + 32 < K)
      __builtin_prefetch((const _Float16*)Ain + (size_t)arow * K + k0 + 32, 0, 0);
    v16h a;
    if (A_F16) a = frag_from_f16((const _Float16*)Ain, K, arow, k0, lane);
    else       a = frag_from_f32((const float*)Ain,    K, arow, k0, lane);

#pragma unroll
    for (int t = 0; t < 4; ++t) {
      const _Float16* bp = &Bt[t * 16 + (lane & 15)][g];
      v8h lo = *(const v8h*)bp;
      v8h hi = *(const v8h*)(bp + 16);
      v16h b = __builtin_shufflevector(lo, hi,
                                       0,1,2,3,4,5,6,7,8,9,10,11,12,13,14,15);
      acc[t] = __builtin_amdgcn_wmma_f32_16x16x32_f16(false, a, false, b,
                                                      (short)0, acc[t], false, false);
    }
  }

  const float scale = fminf(fmaxf(abs_sum[0] * inv_numel, 1e-5f), 1000.0f);
  const int   hi8   = (lane & 16) ? 8 : 0;
#pragma unroll
  for (int t = 0; t < 4; ++t) {
    const int   col  = n0 + t * 16 + (lane & 15);
    const float bcol = bias[col];
#pragma unroll
    for (int r = 0; r < 8; ++r) {            // C/D layout: VGPR r -> M = r (+8)
      int m = m0 + r + hi8;
      if (m >= M) continue;
      int   orow = m + (m / rows_per_batch) * row_pad + row_base;
      float v    = acc[t][r] * scale + bcol;
      if (OUT_MODE == 0) {
        ((float*)Out)[(size_t)orow * out_ld + col] = v;
      } else if (OUT_MODE == 1) {
        ((_Float16*)Out)[(size_t)orow * out_ld + col] = (_Float16)v;
      } else if (OUT_MODE == 2) {            // V: per-head transpose Vt[b][h][d][j]
        int b = orow / LPAD_, j = orow % LPAD_;
        int hh = col >> 6, d = col & 63;
        ((_Float16*)Out)[(((size_t)b * NH_ + hh) * HD_ + d) * LPAD_ + j] = (_Float16)v;
      } else {                               // f32 + f16 shadow
        ((float*)Out)[(size_t)orow * out_ld + col]     = v;
        ((_Float16*)Out2)[(size_t)orow * out_ld + col] = (_Float16)v;
      }
    }
  }
}

// ---------------------------------------------------------------------------
// Flash attention: one wave per (batch, head, 16-row Q tile). Streams keys in
// 32-column steps: 4 WMMAs for S=QK^T, online softmax through LDS, 4 WMMAs for
// O += P*V against pre-transposed Vt (contiguous b128 fragment loads).
// ---------------------------------------------------------------------------
__global__ __launch_bounds__(32) void flash_attn_kernel(
    const _Float16* __restrict__ Q, const _Float16* __restrict__ Kf,
    const _Float16* __restrict__ Vt, const float* __restrict__ attb,
    _Float16* __restrict__ Ctx) {
  __shared__ float    Sld[16][36];   // padded vs bank conflicts
  __shared__ _Float16 Pld[16][32];
  __shared__ float    Alds[16];
  __shared__ float    Llds[16];

  const int lane = threadIdx.x;
  const int mt   = blockIdx.x % MT_;
  const int h    = (blockIdx.x / MT_) % NH_;
  const int b    = blockIdx.x / (MT_ * NH_);
  const size_t base = (size_t)b * LPAD_;
  const _Float16* Qh  = Q  + base * HID_ + h * HD_;
  const _Float16* Kh  = Kf + base * HID_ + h * HD_;
  const _Float16* VtH = Vt + ((size_t)b * NH_ + h) * HD_ * LPAD_;
  const float*    bb  = attb + (size_t)b * LPAD_;
  const int m0  = mt * 16;
  const int hi8 = (lane & 16) ? 8 : 0;
  const int g   = (lane & 16) ? 8 : 0;

  const v16h qf0 = frag_from_f16(Qh, HID_, m0 + (lane & 15), 0, lane);
  const v16h qf1 = frag_from_f16(Qh, HID_, m0 + (lane & 15), 32, lane);

  v8f  o[4] = {};
  float m_i = -1e30f, l_i = 0.f;

  for (int j0 = 0; j0 < LPAD_; j0 += 32) {
    // ---- S tile (16x32) = Q(16x64) . K^T ----
#pragma unroll
    for (int t = 0; t < 2; ++t) {
      const int jrow = j0 + t * 16 + (lane & 15);
      v16h kf0 = frag_from_f16(Kh, HID_, jrow, 0, lane);
      v16h kf1 = frag_from_f16(Kh, HID_, jrow, 32, lane);
      v8f s = {};
      s = __builtin_amdgcn_wmma_f32_16x16x32_f16(false, qf0, false, kf0,
                                                 (short)0, s, false, false);
      s = __builtin_amdgcn_wmma_f32_16x16x32_f16(false, qf1, false, kf1,
                                                 (short)0, s, false, false);
      const int col   = t * 16 + (lane & 15);
      const float biasv = bb[j0 + col];
#pragma unroll
      for (int r = 0; r < 8; ++r) Sld[r + hi8][col] = s[r] * 0.125f + biasv;
    }
    __syncthreads();

    // ---- online softmax: lane i owns row i ----
    if (lane < 16) {
      float rowmax = m_i;
#pragma unroll
      for (int c = 0; c < 32; ++c) rowmax = fmaxf(rowmax, Sld[lane][c]);
      float al = __expf(m_i - rowmax);
      float ls = 0.f;
#pragma unroll
      for (int c = 0; c < 32; ++c) {
        float p = __expf(Sld[lane][c] - rowmax);
        Pld[lane][c] = (_Float16)p;
        ls += p;
      }
      m_i = rowmax;
      l_i = l_i * al + ls;
      Alds[lane] = al;
    }
    __syncthreads();

    // ---- rescale running O ----
    float ar[8];
#pragma unroll
    for (int r = 0; r < 8; ++r) ar[r] = Alds[r + hi8];
#pragma unroll
    for (int t = 0; t < 4; ++t)
#pragma unroll
      for (int r = 0; r < 8; ++r) o[t][r] *= ar[r];

    // ---- P fragment (A operand, 16x32) from LDS ----
    const int prow = lane & 15;
    v8h plo = *(const v8h*)&Pld[prow][g];
    v8h phi = *(const v8h*)&Pld[prow][16 + g];
    v16h pf = __builtin_shufflevector(plo, phi,
                                      0,1,2,3,4,5,6,7,8,9,10,11,12,13,14,15);

    // ---- O += P(16x32) . V(32x64), B frags contiguous from Vt ----
#pragma unroll
    for (int t = 0; t < 4; ++t) {
      v16h vf = frag_from_f16(VtH, LPAD_, t * 16 + (lane & 15), j0, lane);
      o[t] = __builtin_amdgcn_wmma_f32_16x16x32_f16(false, pf, false, vf,
                                                    (short)0, o[t], false, false);
    }
    __syncthreads();
  }

  if (lane < 16) Llds[lane] = (l_i > 0.f) ? 1.0f / l_i : 0.f;
  __syncthreads();
#pragma unroll
  for (int r = 0; r < 8; ++r) {
    float inv = Llds[r + hi8];
    int   row = m0 + r + hi8;
    _Float16* dst = Ctx + (base + row) * HID_ + h * HD_;
#pragma unroll
    for (int t = 0; t < 4; ++t)
      dst[t * 16 + (lane & 15)] = (_Float16)(o[t][r] * inv);
  }
}

// ---------------------------------------------------------------------------
// Residual add + LayerNorm; routes row 0 to vision_output, rows 1..S to
// text_output (d_out layout = outputs concatenated flat).
// ---------------------------------------------------------------------------
__global__ __launch_bounds__(256) void ln_residual_kernel(
    const float* __restrict__ OP, const float* __restrict__ CB,
    const float* __restrict__ gamma, const float* __restrict__ beta,
    float* __restrict__ out) {
  __shared__ float r1[256], r2[256];
  const int row = blockIdx.x;
  const int b = row / L_, l = row % L_;
  const float* xo = OP + ((size_t)b * LPAD_ + l) * HID_;
  const float* xc = CB + ((size_t)b * LPAD_ + l) * HID_;
  float loc[4], s = 0.f, s2 = 0.f;
#pragma unroll
  for (int it = 0; it < 4; ++it) {
    int   c = threadIdx.x + it * 256;
    float v = xo[c] + xc[c];
    loc[it] = v; s += v; s2 += v * v;
  }
  r1[threadIdx.x] = s; r2[threadIdx.x] = s2;
  __syncthreads();
  for (int off = 128; off > 0; off >>= 1) {
    if (threadIdx.x < off) {
      r1[threadIdx.x] += r1[threadIdx.x + off];
      r2[threadIdx.x] += r2[threadIdx.x + off];
    }
    __syncthreads();
  }
  const float mean = r1[0] * (1.0f / HID_);
  const float var  = r2[0] * (1.0f / HID_) - mean * mean;
  const float rstd = rsqrtf(var + 1e-5f);
  float* dst = (l == 0)
                 ? out + (size_t)b * HID_
                 : out + (size_t)B_ * HID_ + ((size_t)b * S_ + (l - 1)) * HID_;
#pragma unroll
  for (int it = 0; it < 4; ++it) {
    int c = threadIdx.x + it * 256;
    dst[c] = (loc[it] - mean) * rstd * gamma[c] + beta[c];
  }
}

// ---------------------------------------------------------------------------
extern "C" void kernel_launch(void* const* d_in, const int* in_sizes, int n_in,
                              void* d_out, int out_size, void* d_ws, size_t ws_size,
                              hipStream_t stream) {
  (void)in_sizes; (void)n_in; (void)out_size; (void)ws_size;
  const float* vis   = (const float*)d_in[0];
  const float* txt   = (const float*)d_in[1];
  const int*   tmask = (const int*)d_in[2];
  const float* W_vis = (const float*)d_in[3];
  const float* b_vis = (const float*)d_in[4];
  const float* W_txt = (const float*)d_in[5];
  const float* b_txt = (const float*)d_in[6];
  const float* W_q   = (const float*)d_in[7];
  const float* b_q   = (const float*)d_in[8];
  const float* W_k   = (const float*)d_in[9];
  const float* b_k   = (const float*)d_in[10];
  const float* W_v   = (const float*)d_in[11];
  const float* b_v   = (const float*)d_in[12];
  const float* W_o   = (const float*)d_in[13];
  const float* b_o   = (const float*)d_in[14];
  const float* g     = (const float*)d_in[15];
  const float* be    = (const float*)d_in[16];

  char*  ws  = (char*)d_ws;
  size_t off = 0;
  auto alloc = [&](size_t bytes) -> char* {
    char* p = ws + off;
    off += (bytes + 255) & ~(size_t)255;
    return p;
  };
  float*    sums   = (float*)alloc(8 * sizeof(float));
  _Float16* qw_vis = (_Float16*)alloc((size_t)HID_ * VDIM_ * 2);
  _Float16* qw_txt = (_Float16*)alloc((size_t)HID_ * TDIM_ * 2);
  _Float16* qw_q   = (_Float16*)alloc((size_t)HID_ * HID_ * 2);
  _Float16* qw_k   = (_Float16*)alloc((size_t)HID_ * HID_ * 2);
  _Float16* qw_v   = (_Float16*)alloc((size_t)HID_ * HID_ * 2);
  _Float16* qw_o   = (_Float16*)alloc((size_t)HID_ * HID_ * 2);
  float*    comb   = (float*)alloc((size_t)B_ * LPAD_ * HID_ * 4);
  _Float16* comb_h = (_Float16*)alloc((size_t)B_ * LPAD_ * HID_ * 2);
  _Float16* Qb     = (_Float16*)alloc((size_t)B_ * LPAD_ * HID_ * 2);
  _Float16* Kb     = (_Float16*)alloc((size_t)B_ * LPAD_ * HID_ * 2);
  _Float16* Vt     = (_Float16*)alloc((size_t)B_ * NH_ * HD_ * LPAD_ * 2);
  _Float16* Cx     = (_Float16*)alloc((size_t)B_ * LPAD_ * HID_ * 2);
  float*    OP     = (float*)alloc((size_t)B_ * LPAD_ * HID_ * 4);
  float*    attb   = (float*)alloc((size_t)B_ * LPAD_ * 4);

  // 1) init scratch + attention bias table
  init_ws_kernel<<<512, 256, 0, stream>>>(sums, comb, comb_h);
  attn_bias_kernel<<<(B_ * LPAD_ + 255) / 256, 256, 0, stream>>>(tmask, attb);

  // 2) mean(|W|) reductions
  abs_sum_kernel<<<256, 256, 0, stream>>>(W_vis, HID_ * VDIM_, sums + 0);
  abs_sum_kernel<<<256, 256, 0, stream>>>(W_txt, HID_ * TDIM_, sums + 1);
  abs_sum_kernel<<<256, 256, 0, stream>>>(W_q,   HID_ * HID_,  sums + 2);
  abs_sum_kernel<<<256, 256, 0, stream>>>(W_k,   HID_ * HID_,  sums + 3);
  abs_sum_kernel<<<256, 256, 0, stream>>>(W_v,   HID_ * HID_,  sums + 4);
  abs_sum_kernel<<<256, 256, 0, stream>>>(W_o,   HID_ * HID_,  sums + 5);

  const float invVD = 1.0f / (float)(HID_ * VDIM_);
  const float invTD = 1.0f / (float)(HID_ * TDIM_);
  const float invHH = 1.0f / (float)(HID_ * HID_);

  // 3) ternary quantization to f16
  quantize_kernel<<<512, 256, 0, stream>>>(W_vis, HID_ * VDIM_, sums + 0, invVD, qw_vis);
  quantize_kernel<<<512, 256, 0, stream>>>(W_txt, HID_ * TDIM_, sums + 1, invTD, qw_txt);
  quantize_kernel<<<512, 256, 0, stream>>>(W_q,   HID_ * HID_,  sums + 2, invHH, qw_q);
  quantize_kernel<<<512, 256, 0, stream>>>(W_k,   HID_ * HID_,  sums + 3, invHH, qw_k);
  quantize_kernel<<<512, 256, 0, stream>>>(W_v,   HID_ * HID_,  sums + 4, invHH, qw_v);
  quantize_kernel<<<512, 256, 0, stream>>>(W_o,   HID_ * HID_,  sums + 5, invHH, qw_o);

  const int BIG = 1 << 30;  // identity row-map
  const int NSTRIPS = HID_ / 64;  // 16

  // 4) vision projection -> combined rows b*LPAD (rpb=1, pad=LPAD-1) [f32+f16]
  gemm_bitlinear<false, 3><<<1 * NSTRIPS, 256, 0, stream>>>(
      vis, qw_vis, b_vis, sums + 0, invVD, comb, comb_h, 16, HID_, VDIM_,
      1, LPAD_ - 1, 0, HID_);
  // 5) text projection -> combined rows b*LPAD + 1 + s [f32+f16]
  gemm_bitlinear<false, 3><<<(1024 / 8) * NSTRIPS, 256, 0, stream>>>(
      txt, qw_txt, b_txt, sums + 1, invTD, comb, comb_h, B_ * S_, HID_, TDIM_,
      S_, LPAD_ - S_, 1, HID_);

  // 6) Q/K/V projections (f16 A from comb_h)
  const int Mqkv = B_ * LPAD_;                 // 16896 rows -> 132 m-blocks
  const int qkv_blocks = (Mqkv / 16 / 8) * NSTRIPS;
  gemm_bitlinear<true, 1><<<qkv_blocks, 256, 0, stream>>>(
      comb_h, qw_q, b_q, sums + 2, invHH, Qb, nullptr, Mqkv, HID_, HID_,
      BIG, 0, 0, HID_);
  gemm_bitlinear<true, 1><<<qkv_blocks, 256, 0, stream>>>(
      comb_h, qw_k, b_k, sums + 3, invHH, Kb, nullptr, Mqkv, HID_, HID_,
      BIG, 0, 0, HID_);
  gemm_bitlinear<true, 2><<<qkv_blocks, 256, 0, stream>>>(
      comb_h, qw_v, b_v, sums + 4, invHH, Vt, nullptr, Mqkv, HID_, HID_,
      BIG, 0, 0, HID_);

  // 7) flash attention -> Ctx (f16)
  flash_attn_kernel<<<B_ * NH_ * MT_, 32, 0, stream>>>(Qb, Kb, Vt, attb, Cx);

  // 8) output projection (f16 A -> f32 out)
  gemm_bitlinear<true, 0><<<qkv_blocks, 256, 0, stream>>>(
      Cx, qw_o, b_o, sums + 5, invHH, OP, nullptr, Mqkv, HID_, HID_,
      BIG, 0, 0, HID_);

  // 9) residual + layernorm -> d_out (vision block, then text block)
  ln_residual_kernel<<<B_ * L_, 256, 0, stream>>>(OP, comb, g, be, (float*)d_out);
}
